// GraphSAGEEncoder_28621662060925
// MI455X (gfx1250) — matrix operand order repaired and checked by reading, not compiled, not run on hardware
//
#include <hip/hip_runtime.h>
#include <hip/hip_bf16.h>

// CDNA5 / gfx1250, wave32. Matrix math via v_wmma_f32_16x16x32_f16.

typedef __attribute__((ext_vector_type(16))) _Float16 v16h;
typedef __attribute__((ext_vector_type(8)))  float    v8f;

#define N_NODES 100000
#define N_EDGES 1600000
#define IN_DIM  128
#define HID_DIM 64
#define OUT_DIM 128

// ---------------------------------------------------------------- zero fill
__global__ void zero_f32(float* __restrict__ p, long n) {
    long i      = (long)blockIdx.x * blockDim.x + threadIdx.x;
    long stride = (long)gridDim.x * blockDim.x;
    for (; i < n; i += stride) p[i] = 0.0f;
}

// ---------------------------------------------------------------- in-degree
__global__ void deg_count(const int* __restrict__ dst, float* __restrict__ deg, int E) {
    int e = blockIdx.x * blockDim.x + threadIdx.x;
    if (e < E) atomicAdd(&deg[dst[e]], 1.0f);
}

// ------------------------------------------------- edge feature scatter-add
// One wave (32 lanes) per edge; lane handles VEC contiguous floats so the
// wave reads a full 32*VEC*4-byte row coalesced and issues contiguous
// global_atomic_add_f32 (resolved in the 192MB L2 -- whole graph fits).
template <int VEC>
__global__ void scatter_feats(const float* __restrict__ x,
                              const int* __restrict__ src,
                              const int* __restrict__ dst,
                              float* __restrict__ aggr, int E) {
    int gid  = blockIdx.x * blockDim.x + threadIdx.x;
    int e    = gid >> 5;
    int lane = gid & 31;
    if (e >= E) return;
    const int D = 32 * VEC;
    int s = src[e];
    int d = dst[e];
    const float* xr = x    + (long)s * D + lane * VEC;
    float*       ar = aggr + (long)d * D + lane * VEC;
    float v[VEC];
#pragma unroll
    for (int j = 0; j < VEC; ++j) v[j] = xr[j];      // merges to b128/b64 load
#pragma unroll
    for (int j = 0; j < VEC; ++j) atomicAdd(&ar[j], v[j]);
}

// ------------------------------------------------------ WMMA operand packing
// A-matrix 16x32 f16 layout (ISA 7.12.2): lanes 0-15 and 16-31 both hold
// row M = lane&15; low half-wave carries K = {hi*8+0..7, 16+hi*8+0..7}.
__device__ __forceinline__ v16h pack_a(const float* __restrict__ row,
                                       int kbase, int hi, float scale) {
    v16h a;
    int kb = kbase + hi * 8;
#pragma unroll
    for (int j = 0; j < 8; ++j) a[j]     = (_Float16)(row[kb + j]      * scale);
#pragma unroll
    for (int j = 0; j < 8; ++j) a[8 + j] = (_Float16)(row[kb + 16 + j] * scale);
    return a;
}

// B-matrix 32x16 f16: lane = column N; half-wave hi carries
// K = {hi*8+0..7, 16+hi*8+0..7} (same order as A).
__device__ __forceinline__ v16h pack_b(const float* __restrict__ wrow,
                                       int kbase, int hi) {
    v16h b;
    int kb = kbase + hi * 8;
#pragma unroll
    for (int j = 0; j < 8; ++j) b[j]     = (_Float16)wrow[kb + j];
#pragma unroll
    for (int j = 0; j < 8; ++j) b[8 + j] = (_Float16)wrow[kb + 16 + j];
    return b;
}

// --------------------------------------------------------- fused SAGE layer
// out = [relu]( (aggr/deg) @ Wl^T + bias + self @ Wr^T )
// One wave per 16-node tile. KDIM, NDIM multiples of 32/16; N_NODES % 16 == 0
// so EXEC is all-ones around every WMMA (required by ISA).
template <int KDIM, int NDIM, bool RELU>
__global__ void __launch_bounds__(32)
sage_layer(const float* __restrict__ selfF, const float* __restrict__ aggrF,
           const float* __restrict__ deg,
           const float* __restrict__ Wl, const float* __restrict__ Wr,
           const float* __restrict__ bias, float* __restrict__ out) {
    constexpr int KT = KDIM / 32;   // K-steps of 32
    constexpr int NT = NDIM / 16;   // 16-wide output tiles

    int lane = threadIdx.x;
    int r    = lane & 15;
    int hi   = lane >> 4;
    long nodeBase = (long)blockIdx.x * 16;
    long arow     = nodeBase + r;

    float inv = 1.0f / fmaxf(deg[arow], 1.0f);   // mean = sum * inv (0 if deg==0)
    const float* selfRow = selfF + arow * KDIM;
    const float* aggrRow = aggrF + arow * KDIM;

    v16h a_self[KT], a_aggr[KT];
#pragma unroll
    for (int kt = 0; kt < KT; ++kt) {
        a_self[kt] = pack_a(selfRow, kt * 32, hi, 1.0f);
        a_aggr[kt] = pack_a(aggrRow, kt * 32, hi, inv);  // fold 1/deg into f16 cvt
    }

#pragma unroll
    for (int nt = 0; nt < NT; ++nt) {
        int n = nt * 16 + r;                     // weight row this lane supplies
        const float* wl = Wl + (long)n * KDIM;
        const float* wr = Wr + (long)n * KDIM;
        v8f acc = {};
#pragma unroll
        for (int kt = 0; kt < KT; ++kt) {
            v16h bl = pack_b(wl, kt * 32, hi);
            v16h br = pack_b(wr, kt * 32, hi);
            acc = __builtin_amdgcn_wmma_f32_16x16x32_f16(
                false, a_aggr[kt], false, bl, (short)0, acc, false, false);
            acc = __builtin_amdgcn_wmma_f32_16x16x32_f16(
                false, a_self[kt], false, br, (short)0, acc, false, false);
        }
        // C/D layout: element v of acc -> (M = hi*8 + v, N = r)
        float bv = bias[nt * 16 + r];
#pragma unroll
        for (int v = 0; v < 8; ++v) {
            float val = acc[v] + bv;
            if (RELU) val = fmaxf(val, 0.0f);
            int m = hi * 8 + v;
            out[(nodeBase + m) * NDIM + nt * 16 + r] = val;
        }
    }
}

// ------------------------------------------------------------------ launcher
extern "C" void kernel_launch(void* const* d_in, const int* in_sizes, int n_in,
                              void* d_out, int out_size, void* d_ws, size_t ws_size,
                              hipStream_t stream) {
    const float* x   = (const float*)d_in[0];
    const int*   ei  = (const int*)  d_in[1];   // [2, E]
    const float* Wl1 = (const float*)d_in[2];
    const float* Wr1 = (const float*)d_in[3];
    const float* b1  = (const float*)d_in[4];
    const float* Wl2 = (const float*)d_in[5];
    const float* Wr2 = (const float*)d_in[6];
    const float* b2  = (const float*)d_in[7];
    const int* src = ei;
    const int* dst = ei + N_EDGES;

    // workspace: deg | aggr1 (reused as aggr2) | h     (~77 MB total)
    float* deg   = (float*)d_ws;
    float* aggr1 = deg + N_NODES;
    float* h     = aggr1 + (long)N_NODES * IN_DIM;
    float* aggr2 = aggr1;
    float* out   = (float*)d_out;

    // --- layer 1 ---
    zero_f32<<<2048, 256, 0, stream>>>(deg, (long)N_NODES * (1 + IN_DIM));
    deg_count<<<(N_EDGES + 255) / 256, 256, 0, stream>>>(dst, deg, N_EDGES);
    scatter_feats<4><<<(N_EDGES * 32 + 255) / 256, 256, 0, stream>>>(
        x, src, dst, aggr1, N_EDGES);
    sage_layer<IN_DIM, HID_DIM, true><<<N_NODES / 16, 32, 0, stream>>>(
        x, aggr1, deg, Wl1, Wr1, b1, h);

    // --- layer 2 ---
    zero_f32<<<2048, 256, 0, stream>>>(aggr2, (long)N_NODES * HID_DIM);
    scatter_feats<2><<<(N_EDGES * 32 + 255) / 256, 256, 0, stream>>>(
        h, src, dst, aggr2, N_EDGES);
    sage_layer<HID_DIM, OUT_DIM, false><<<N_NODES / 16, 32, 0, stream>>>(
        h, aggr2, deg, Wl2, Wr2, b2, out);
}
